// SFFN_73409581023885
// MI455X (gfx1250) — matrix operand order repaired and checked by reading, not compiled, not run on hardware
//
#include <hip/hip_runtime.h>

// Problem constants (from reference)
#define Bc   4
#define Tc   4096
#define Hc   8
#define LDc  128
#define FFNc 512
#define Dc   1024   // H*LD
#define MT   32     // tokens per block (two 16-row M-subtiles)

typedef __attribute__((ext_vector_type(16))) __bf16 bf16x16;
typedef __attribute__((ext_vector_type(8)))  float  f32x8;

// ---------------------------------------------------------------------------
// Fragment loaders for V_WMMA_F32_16X16X32_BF16 (wave32).
// A (16xK row-major, per-lane row M=lane%16):
//   frag[0..7]  = K[(hi?8:0) + 0..7],  frag[8..15] = K[16+(hi?8:0) + 0..7]
// B (Kx16; we store W^T row-major so a "row" is a column of W, per-lane
//   col N=lane%16): same K order as A.
// ---------------------------------------------------------------------------
__device__ __forceinline__ bf16x16 load_a_frag(const __bf16* p, int hi) {
    union { bf16x16 v; uint4 q[2]; } u;
    const char* c = (const char*)p + hi * 16;
    u.q[0] = *(const uint4*)(c);
    u.q[1] = *(const uint4*)(c + 32);
    return u.v;
}
__device__ __forceinline__ bf16x16 load_b_frag(const __bf16* p, int hi) {
    union { bf16x16 v; uint4 q[2]; } u;
    const char* c = (const char*)p + hi * 16;
    u.q[0] = *(const uint4*)(c);
    u.q[1] = *(const uint4*)(c + 32);
    return u.v;
}

__device__ __forceinline__ float gelu_exact(float v) {
    return 0.5f * v * (1.0f + erff(v * 0.70710678118654752f));
}
__device__ __forceinline__ float silu(float v) {
    return v / (1.0f + expf(-v));
}

// ---------------------------------------------------------------------------
// Weight conversion kernels: f32 -> bf16, transposed for B-fragment loads.
// ---------------------------------------------------------------------------
__global__ void cvt_w1(const float* __restrict__ W1, __bf16* __restrict__ W1T) {
    // W1 (H,LD,FFN) -> W1T (H,FFN,LD)
    int idx = blockIdx.x * 256 + threadIdx.x;          // < H*FFN*LD = 524288
    int d = idx & 127, f = (idx >> 7) & 511, h = idx >> 16;
    W1T[idx] = (__bf16)W1[(h * LDc + d) * FFNc + f];
}
__global__ void cvt_w2(const float* __restrict__ W2, __bf16* __restrict__ W2T) {
    // W2 (H,FFN,LD) -> W2T (H,LD,FFN)
    int idx = blockIdx.x * 256 + threadIdx.x;          // < H*LD*FFN = 524288
    int f = idx & 511, d = (idx >> 9) & 127, h = idx >> 16;
    W2T[idx] = (__bf16)W2[(h * FFNc + f) * LDc + d];
}
__global__ void cvt_ow(const float* __restrict__ oW, __bf16* __restrict__ oWT) {
    // out_W (D,D)[k][n] -> oWT[n][k]
    int idx = blockIdx.x * 256 + threadIdx.x;          // < D*D = 1048576
    int k = idx & 1023, n = idx >> 10;
    oWT[idx] = (__bf16)oW[k * Dc + n];
}

// ---------------------------------------------------------------------------
// Stylization emb path: ss[b][0:2D] = silu(emb[b]) @ emb_W + emb_b  (tiny)
// ---------------------------------------------------------------------------
__global__ void emb_gemm(const float* __restrict__ emb,
                         const float* __restrict__ embW,
                         const float* __restrict__ embB,
                         float* __restrict__ ss) {
    int idx = blockIdx.x * 256 + threadIdx.x;          // < B*2D = 8192
    int b = idx >> 11, n = idx & 2047;
    float acc = embB[n];
    const float* e = emb + b * 2048;
    #pragma unroll 4
    for (int k = 0; k < 2048; ++k) {
        float v = e[k];
        acc += (v / (1.0f + expf(-v))) * embW[(size_t)k * 2048 + n];
    }
    ss[idx] = acc;
}

// ---------------------------------------------------------------------------
// Fused: per-head MLP (WMMA) -> LN (register-resident) -> modulate -> SiLU
//        -> out GEMM (WMMA). 1 block = 32 tokens, 8 waves, wave w == head w.
// Every B fragment (weight) is reused for 2 M-subtiles -> halved L2 traffic.
// Dynamic LDS: [0,64K)   x tile bf16 [32][1024], reused as h bf16
//              [64K,128K) per-head h1 chunks bf16 [8][32][128]
//              [128K, ..) LN partial sums + mu/rstd
// ---------------------------------------------------------------------------
__global__ __launch_bounds__(256) void fused_mlp_ln_out(
    const float* __restrict__ x,
    const float* __restrict__ b1,
    const float* __restrict__ b2,
    const float* __restrict__ ln_w,
    const float* __restrict__ ln_b,
    const float* __restrict__ out_b,
    const __bf16* __restrict__ W1T,
    const __bf16* __restrict__ W2T,
    const __bf16* __restrict__ oWT,
    const float* __restrict__ ss,
    float* __restrict__ out)
{
    extern __shared__ char smem[];
    __bf16* lds_xh  = (__bf16*)smem;                 // [32][1024] bf16 (x, then h)
    __bf16* lds_h1  = (__bf16*)(smem + 65536);       // [8][32][128] bf16
    float*  lds_sum = (float*)(smem + 131072);       // [32][8]
    float*  lds_sq  = lds_sum + 256;                 // [32][8]
    float*  lds_mu  = lds_sq + 256;                  // [32]
    float*  lds_rs  = lds_mu + 32;                   // [32]

    const int token0 = blockIdx.x * MT;              // flat token in (B*T)
    const int b      = token0 / Tc;
    const int tid    = threadIdx.x;
    const int h      = tid >> 5;                     // wave == head
    const int lane   = tid & 31;
    const int l16    = lane & 15;
    const int hi     = lane >> 4;

    // ---- load x tile (32 x 1024 f32) -> bf16 LDS ----
    {
        const float4* xg = (const float4*)(x + (size_t)token0 * Dc);
        for (int i = tid; i < MT * Dc / 4; i += 256) {
            float4 v = xg[i];
            __bf16* d = lds_xh + i * 4;
            d[0] = (__bf16)v.x; d[1] = (__bf16)v.y;
            d[2] = (__bf16)v.z; d[3] = (__bf16)v.w;
        }
    }
    __syncthreads();

    // ---- A fragments of this head's x slice (2 x 16 x 128) ----
    bf16x16 a1[2][4];
    #pragma unroll
    for (int ms = 0; ms < 2; ++ms)
        #pragma unroll
        for (int kc = 0; kc < 4; ++kc)
            a1[ms][kc] = load_a_frag(
                lds_xh + (ms * 16 + l16) * Dc + h * LDc + kc * 32, hi);

    f32x8 c2[2][8];
    #pragma unroll
    for (int ms = 0; ms < 2; ++ms)
        #pragma unroll
        for (int i = 0; i < 8; ++i) c2[ms][i] = (f32x8)0.0f;

    // ---- MLP: K-chunked over FFN; h1 chunk lives in 8KB/head LDS ----
    #pragma unroll 1
    for (int fc = 0; fc < 4; ++fc) {
        // GEMM1 chunk: (32 x 128) of h1 = gelu(x @ W1 + b1)
        #pragma unroll
        for (int nt = 0; nt < 8; ++nt) {
            const int f = fc * 128 + nt * 16 + l16;
            const __bf16* wrow = W1T + (size_t)(h * FFNc + f) * LDc;
            bf16x16 bb[4];
            #pragma unroll
            for (int kc = 0; kc < 4; ++kc) bb[kc] = load_b_frag(wrow + kc * 32, hi);

            f32x8 acc0 = (f32x8)0.0f, acc1 = (f32x8)0.0f;
            #pragma unroll
            for (int kc = 0; kc < 4; ++kc) {
                acc0 = __builtin_amdgcn_wmma_f32_16x16x32_bf16(
                           false, a1[0][kc], false, bb[kc], (short)0, acc0, false, false);
                acc1 = __builtin_amdgcn_wmma_f32_16x16x32_bf16(
                           false, a1[1][kc], false, bb[kc], (short)0, acc1, false, false);
            }
            const float bias = b1[h * FFNc + f];
            #pragma unroll
            for (int r = 0; r < 8; ++r) {
                lds_h1[h * 4096 + (r + hi * 8) * 128 + nt * 16 + l16] =
                    (__bf16)gelu_exact(acc0[r] + bias);
                lds_h1[h * 4096 + (16 + r + hi * 8) * 128 + nt * 16 + l16] =
                    (__bf16)gelu_exact(acc1[r] + bias);
            }
        }
        // wave-private LDS RAW: stores above, fragment loads below
        asm volatile("s_wait_dscnt 0" ::: "memory");

        bf16x16 a2[2][4];
        #pragma unroll
        for (int ms = 0; ms < 2; ++ms)
            #pragma unroll
            for (int kc = 0; kc < 4; ++kc)
                a2[ms][kc] = load_a_frag(
                    lds_h1 + h * 4096 + (ms * 16 + l16) * 128 + kc * 32, hi);

        // GEMM2 partial accumulate: y += h1_chunk @ W2[fc*128 : ...]
        #pragma unroll
        for (int nt = 0; nt < 8; ++nt) {
            const int d = nt * 16 + l16;
            const __bf16* wrow = W2T + (size_t)(h * LDc + d) * FFNc + fc * 128;
            #pragma unroll
            for (int kc = 0; kc < 4; ++kc) {
                bf16x16 bb = load_b_frag(wrow + kc * 32, hi);
                c2[0][nt] = __builtin_amdgcn_wmma_f32_16x16x32_bf16(
                                false, a2[0][kc], false, bb, (short)0, c2[0][nt], false, false);
                c2[1][nt] = __builtin_amdgcn_wmma_f32_16x16x32_bf16(
                                false, a2[1][kc], false, bb, (short)0, c2[1][nt], false, false);
            }
        }
    }

    // ---- add b2 into register-resident y ----
    #pragma unroll
    for (int nt = 0; nt < 8; ++nt) {
        const float bias = b2[h * LDc + nt * 16 + l16];
        #pragma unroll
        for (int r = 0; r < 8; ++r) {
            c2[0][nt][r] += bias;
            c2[1][nt][r] += bias;
        }
    }

    // ---- LN stats from registers: per-row partial sum/sumsq over this
    //      head's 128 columns, reduced across the 16-lane group ----
    #pragma unroll
    for (int ms = 0; ms < 2; ++ms) {
        float p[8], q[8];
        #pragma unroll
        for (int r = 0; r < 8; ++r) {
            float s = 0.f, s2 = 0.f;
            #pragma unroll
            for (int nt = 0; nt < 8; ++nt) {
                float v = c2[ms][nt][r];
                s += v; s2 += v * v;
            }
            p[r] = s; q[r] = s2;
        }
        #pragma unroll
        for (int m = 8; m >= 1; m >>= 1)
            #pragma unroll
            for (int r = 0; r < 8; ++r) {
                p[r] += __shfl_xor(p[r], m, 32);
                q[r] += __shfl_xor(q[r], m, 32);
            }
        if (l16 == 0) {
            #pragma unroll
            for (int r = 0; r < 8; ++r) {
                const int row = ms * 16 + r + hi * 8;
                lds_sum[row * 8 + h] = p[r];
                lds_sq [row * 8 + h] = q[r];
            }
        }
    }
    __syncthreads();

    // ---- wave 0 finalizes mu / rstd for all 32 rows ----
    if (h == 0) {
        const int row = lane;
        float s = 0.f, s2 = 0.f;
        #pragma unroll
        for (int j = 0; j < 8; ++j) { s += lds_sum[row * 8 + j]; s2 += lds_sq[row * 8 + j]; }
        const float mu  = s * (1.0f / Dc);
        const float var = s2 * (1.0f / Dc) - mu * mu;
        lds_mu[row] = mu;
        lds_rs[row] = rsqrtf(var + 1e-5f);
    }
    __syncthreads();

    // ---- modulate + SiLU in registers -> bf16 h into (dead) x buffer ----
    const float* ssb = ss + (size_t)b * 2 * Dc;
    #pragma unroll
    for (int nt = 0; nt < 8; ++nt) {
        const int col = h * LDc + nt * 16 + l16;
        const float lw = ln_w[col], lb = ln_b[col];
        const float sc = 1.0f + ssb[col], sh = ssb[Dc + col];
        #pragma unroll
        for (int ms = 0; ms < 2; ++ms)
            #pragma unroll
            for (int r = 0; r < 8; ++r) {
                const int row = ms * 16 + r + hi * 8;
                float yn = (c2[ms][nt][r] - lds_mu[row]) * lds_rs[row] * lw + lb;
                lds_xh[row * Dc + col] = (__bf16)silu(yn * sc + sh);
            }
    }
    __syncthreads();

    // ---- out GEMM: (32 x 1024) = h @ out_W; wave h owns N in
    //      [h*128, h*128+128); K = 1024 in 32 chunks; B reused for 2 M ----
    f32x8 co[2][8];
    #pragma unroll
    for (int ms = 0; ms < 2; ++ms)
        #pragma unroll
        for (int i = 0; i < 8; ++i) co[ms][i] = (f32x8)0.0f;

    #pragma unroll 1
    for (int kc = 0; kc < 32; ++kc) {
        bf16x16 aa0 = load_a_frag(lds_xh + l16 * Dc + kc * 32, hi);
        bf16x16 aa1 = load_a_frag(lds_xh + (16 + l16) * Dc + kc * 32, hi);
        #pragma unroll
        for (int nt = 0; nt < 8; ++nt) {
            const int n = h * 128 + nt * 16 + l16;
            bf16x16 bb = load_b_frag(oWT + (size_t)n * Dc + kc * 32, hi);
            co[0][nt] = __builtin_amdgcn_wmma_f32_16x16x32_bf16(
                            false, aa0, false, bb, (short)0, co[0][nt], false, false);
            co[1][nt] = __builtin_amdgcn_wmma_f32_16x16x32_bf16(
                            false, aa1, false, bb, (short)0, co[1][nt], false, false);
        }
    }

    // ---- epilogue: + out_b + residual x ----
    #pragma unroll
    for (int nt = 0; nt < 8; ++nt) {
        const int n = h * 128 + nt * 16 + l16;
        const float bias = out_b[n];
        #pragma unroll
        for (int ms = 0; ms < 2; ++ms)
            #pragma unroll
            for (int r = 0; r < 8; ++r) {
                const int row = ms * 16 + r + hi * 8;
                const size_t g = (size_t)(token0 + row) * Dc + n;
                out[g] = co[ms][nt][r] + bias + x[g];
            }
    }
}

// ---------------------------------------------------------------------------
extern "C" void kernel_launch(void* const* d_in, const int* in_sizes, int n_in,
                              void* d_out, int out_size, void* d_ws, size_t ws_size,
                              hipStream_t stream) {
    const float* x    = (const float*)d_in[0];
    const float* emb  = (const float*)d_in[1];
    const float* W1   = (const float*)d_in[2];
    const float* b1   = (const float*)d_in[3];
    const float* W2   = (const float*)d_in[4];
    const float* b2   = (const float*)d_in[5];
    const float* ln_w = (const float*)d_in[6];
    const float* ln_b = (const float*)d_in[7];
    const float* embW = (const float*)d_in[8];
    const float* embB = (const float*)d_in[9];
    const float* outW = (const float*)d_in[10];
    const float* outB = (const float*)d_in[11];
    float* out = (float*)d_out;

    // Workspace layout: W1T(1MB) | W2T(1MB) | oWT(2MB) | ss(32KB)
    char* ws = (char*)d_ws;
    __bf16* W1T = (__bf16*)(ws);
    __bf16* W2T = (__bf16*)(ws + (1u << 20));
    __bf16* oWT = (__bf16*)(ws + (2u << 20));
    float*  ss  = (float*)(ws + (4u << 20));

    cvt_w1<<<2048, 256, 0, stream>>>(W1, W1T);
    cvt_w2<<<2048, 256, 0, stream>>>(W2, W2T);
    cvt_ow<<<4096, 256, 0, stream>>>(outW, oWT);
    emb_gemm<<<32, 256, 0, stream>>>(emb, embW, embB, ss);

    const int smem = 131072 + 2048 + 256;  // 130.25 KB dynamic LDS
    hipFuncSetAttribute((const void*)fused_mlp_ln_out,
                        hipFuncAttributeMaxDynamicSharedMemorySize, smem);

    const int tiles = (Bc * Tc) / MT;  // 512 blocks
    fused_mlp_ln_out<<<tiles, 256, smem, stream>>>(
        x, b1, b2, ln_w, ln_b, outB, W1T, W2T, oWT, ss, out);
}
